// GotenInteractionLayer_14791867367988
// MI455X (gfx1250) — matrix operand order repaired and checked by reading, not compiled, not run on hardware
//
#include <hip/hip_runtime.h>
#include <hip/hip_bf16.h>

#define NN 8192
#define EE 65536
#define LAT 128
#define GUSED 816           // used columns of the 944-wide GEMM
#define ENVT 51             // 816/16 column tiles
#define QT 16               // 256/16 column tiles
#define TOTT (ENVT + QT)    // 67
#define EPSV 1e-5f

typedef __bf16 v16bf __attribute__((ext_vector_type(16)));
typedef float  v8f   __attribute__((ext_vector_type(8)));

__device__ __forceinline__ void atomicMaxFloat(float* addr, float val) {
    if (val >= 0.0f) atomicMax((int*)addr, __float_as_int(val));
    else             atomicMin((unsigned int*)addr, __float_as_uint(val));
}

// ---- convert fp32 row-major weight [128 x row_stride] -> bf16 col-major [ncols x 128]
__global__ __launch_bounds__(256) void conv_w(const float* __restrict__ W,
                                              __bf16* __restrict__ out,
                                              int ncols, int row_stride) {
    int i = blockIdx.x * 256 + threadIdx.x;
    if (i >= ncols * 128) return;
    int c = i >> 7, k = i & 127;
    out[(size_t)c * 128 + k] = (__bf16)W[(size_t)k * row_stride + c];
}

__global__ __launch_bounds__(256) void init_max(float* p, int n) {
    int i = blockIdx.x * 256 + threadIdx.x;
    if (i < n) p[i] = -3.0e38f;
}

// ---- per-node: g = silu(h@W1)@W2 (816 cols), Knode = h@W_key (256 cols)
__global__ __launch_bounds__(256)
void node_kernel(const float* __restrict__ h,
                 const __bf16* __restrict__ W1c,   // [128][128] col-major
                 const __bf16* __restrict__ W2c,   // [816][128]
                 const __bf16* __restrict__ Wkc,   // [256][128]
                 float* __restrict__ g,            // [NN][816]
                 float* __restrict__ Knode) {      // [NN][256]
    __shared__ __bf16 sH[16][128];
    __shared__ __bf16 sS[16][128];
    const int tid = threadIdx.x;
    const int wave = tid >> 5, lane = tid & 31;
    const int half = lane >> 4, m = lane & 15, n = lane & 15;
    const int tileN = blockIdx.x * 16;

    for (int i = tid; i < 16 * 128; i += 256) {
        int r = i >> 7, c = i & 127;
        sH[r][c] = (__bf16)h[(size_t)(tileN + r) * 128 + c];
    }
    __syncthreads();

    // A fragments from h
    v16bf aH[4];
#pragma unroll
    for (int ks = 0; ks < 4; ++ks) {
        int k0 = ks * 32;
#pragma unroll
        for (int e = 0; e < 8; ++e) aH[ks][e]     = sH[m][k0 + 8 * half + e];
#pragma unroll
        for (int e = 0; e < 8; ++e) aH[ks][8 + e] = sH[m][k0 + 16 + 8 * half + e];
    }

    // phase 1: s = silu(h@W1), one 16-col tile per wave (8 tiles total)
    {
        int ct = wave;
        v8f d;
#pragma unroll
        for (int r = 0; r < 8; ++r) d[r] = 0.0f;
#pragma unroll
        for (int ks = 0; ks < 4; ++ks) {
            v16bf b;
            const __bf16* bp = W1c + ((size_t)(ct * 16 + n)) * 128 + ks * 32;
#pragma unroll
            for (int e = 0; e < 8; ++e) { b[e] = bp[8 * half + e]; b[8 + e] = bp[16 + 8 * half + e]; }
            d = __builtin_amdgcn_wmma_f32_16x16x32_bf16(false, aH[ks], false, b, (short)0, d, false, false);
        }
#pragma unroll
        for (int r = 0; r < 8; ++r) {
            int mm = r + 8 * half;
            float v = d[r];
            float sv = v * __builtin_amdgcn_rcpf(1.0f + __expf(-v));   // silu via v_rcp_f32
            sS[mm][ct * 16 + n] = (__bf16)sv;
        }
    }
    __syncthreads();

    v16bf aS[4];
#pragma unroll
    for (int ks = 0; ks < 4; ++ks) {
        int k0 = ks * 32;
#pragma unroll
        for (int e = 0; e < 8; ++e) aS[ks][e]     = sS[m][k0 + 8 * half + e];
#pragma unroll
        for (int e = 0; e < 8; ++e) aS[ks][8 + e] = sS[m][k0 + 16 + 8 * half + e];
    }

    // phase 2: tiles 0..50 -> g (A=sS, B=W2c); tiles 51..66 -> Knode (A=sH, B=Wkc)
    for (int t = wave; t < TOTT; t += 8) {
        bool isK = (t >= ENVT);
        int ct = isK ? (t - ENVT) : t;
        const __bf16* Bc = isK ? Wkc : W2c;
        v8f d;
#pragma unroll
        for (int r = 0; r < 8; ++r) d[r] = 0.0f;
#pragma unroll
        for (int ks = 0; ks < 4; ++ks) {
            v16bf b;
            const __bf16* bp = Bc + ((size_t)(ct * 16 + n)) * 128 + ks * 32;
#pragma unroll
            for (int e = 0; e < 8; ++e) { b[e] = bp[8 * half + e]; b[8 + e] = bp[16 + 8 * half + e]; }
            d = __builtin_amdgcn_wmma_f32_16x16x32_bf16(false, isK ? aH[ks] : aS[ks],
                                                        false, b, (short)0, d, false, false);
        }
#pragma unroll
        for (int r = 0; r < 8; ++r) {
            int mm = r + 8 * half;
            if (isK) Knode[(size_t)(tileN + mm) * 256 + ct * 16 + n] = d[r];
            else     g[(size_t)(tileN + mm) * GUSED + ct * 16 + n] = d[r];
        }
    }
}

// ---- per-edge heavy kernel: env=(t@Wrs)*g_j, delta_h atomics, x=spharm+eqlin, Q, scores+max
__global__ __launch_bounds__(256)
void edge_kernel(const float* __restrict__ t_ij,
                 const float* __restrict__ spharms,
                 const float* __restrict__ Xg,          // [NN][144]
                 const int* __restrict__ ecen,
                 const int* __restrict__ enbr,
                 const __bf16* __restrict__ Wrsc,       // [816][128]
                 const __bf16* __restrict__ Wqc,        // [256][128]
                 const float* __restrict__ g,           // [NN][816]
                 const float* __restrict__ Knode,       // [NN][256]
                 float* __restrict__ delta_h,           // [NN][128]
                 float* __restrict__ maxbuf,            // [NN][16]
                 float* __restrict__ scores,            // [EE][16]
                 float* __restrict__ xbuf) {            // [EE][144]
    __shared__ __bf16 sA[16][128];
    __shared__ float  sEnv[16][GUSED];
    __shared__ float  sQ[16][256];
    __shared__ float  sXj[16][144];
    __shared__ float  sSph[16][9];
    __shared__ int    sCen[16], sNbr[16];

    const int tid = threadIdx.x;
    const int wave = tid >> 5, lane = tid & 31;
    const int half = lane >> 4, m = lane & 15, n = lane & 15;
    const int tileE = blockIdx.x * 16;

    if (tid < 16) { sCen[tid] = ecen[tileE + tid]; sNbr[tid] = enbr[tileE + tid]; }
    for (int i = tid; i < 16 * 128; i += 256) {
        int r = i >> 7, c = i & 127;
        sA[r][c] = (__bf16)t_ij[(size_t)(tileE + r) * 128 + c];
    }
    if (tid < 144) {
        int r = tid / 9, s = tid % 9;
        sSph[r][s] = spharms[(size_t)(tileE + r) * 9 + s];
    }
    __syncthreads();

    // gather X_j tile (needs sNbr)
    for (int i = tid; i < 16 * 144; i += 256) {
        int r = i / 144, c = i % 144;
        sXj[r][c] = Xg[(size_t)sNbr[r] * 144 + c];
    }

    // A fragments (shared by all column tiles of this wave)
    v16bf a[4];
#pragma unroll
    for (int ks = 0; ks < 4; ++ks) {
        int k0 = ks * 32;
#pragma unroll
        for (int e = 0; e < 8; ++e) a[ks][e]     = sA[m][k0 + 8 * half + e];
#pragma unroll
        for (int e = 0; e < 8; ++e) a[ks][8 + e] = sA[m][k0 + 16 + 8 * half + e];
    }

    // 51 env tiles + 16 Q tiles over 8 waves
    for (int t = wave; t < TOTT; t += 8) {
        bool isQ = (t >= ENVT);
        int ct = isQ ? (t - ENVT) : t;
        const __bf16* Bc = isQ ? Wqc : Wrsc;
        v8f d;
#pragma unroll
        for (int r = 0; r < 8; ++r) d[r] = 0.0f;
#pragma unroll
        for (int ks = 0; ks < 4; ++ks) {
            v16bf b;
            const __bf16* bp = Bc + ((size_t)(ct * 16 + n)) * 128 + ks * 32;
#pragma unroll
            for (int e = 0; e < 8; ++e) { b[e] = bp[8 * half + e]; b[8 + e] = bp[16 + 8 * half + e]; }
            d = __builtin_amdgcn_wmma_f32_16x16x32_bf16(false, a[ks], false, b, (short)0, d, false, false);
        }
#pragma unroll
        for (int r = 0; r < 8; ++r) {
            int mm = r + 8 * half;
            int col = ct * 16 + n;
            if (isQ) {
                sQ[mm][col] = d[r];
            } else {
                float gv = g[(size_t)sNbr[mm] * GUSED + col];
                sEnv[mm][col] = d[r] * gv;
            }
        }
    }
    __syncthreads();

    // delta_h scatter (cols 0..127)
    for (int i = tid; i < 16 * 128; i += 256) {
        int r = i >> 7, c = i & 127;
        atomicAdd(&delta_h[(size_t)sCen[r] * 128 + c], sEnv[r][c]);
    }

    // x = env_weighter + eq_linear; 16 edges * 16 mult * 9 sph = 2304
    for (int i = tid; i < 2304; i += 256) {
        int r = i / 144, rem = i % 144;
        int u = rem / 9, s = rem % 9;
        int irr = (s == 0) ? 0 : (s < 4 ? 1 : 2);
        float val = sSph[r][s] * sEnv[r][irr * 16 + u];
        const float* lw = &sEnv[r][48 + irr * 256 + u * 16];
        float acc = 0.0f;
#pragma unroll
        for (int v = 0; v < 16; ++v) acc += lw[v] * sXj[r][v * 9 + s];
        val += 0.25f * acc;                        // 1/sqrt(MULT)
        xbuf[(size_t)(tileE + r) * 144 + u * 9 + s] = val;
    }

    // scores: one (edge,mult) per thread
    {
        int r = tid >> 4, u = tid & 15;
        const float* kp = &Knode[(size_t)sNbr[r] * 256 + u * 16];
        float sc = 0.0f;
#pragma unroll
        for (int hh = 0; hh < 16; ++hh) sc += sQ[r][u * 16 + hh] * kp[hh];
        sc *= 4.0f;                                // ISQRTD
        scores[(size_t)(tileE + r) * 16 + u] = sc;
        atomicMaxFloat(&maxbuf[(size_t)sCen[r] * 16 + u], sc);
    }
}

// ---- softmax pass 2: ex = exp(score - max), accumulate denominator
__global__ __launch_bounds__(256)
void ex_kernel(const int* __restrict__ ecen, const float* __restrict__ maxbuf,
               float* __restrict__ scores, float* __restrict__ denom) {
    int i = blockIdx.x * 256 + threadIdx.x;
    if (i >= EE * 16) return;
    int e = i >> 4, u = i & 15;
    int c = ecen[e];
    float mv = maxbuf[c * 16 + u];
    float ex = __expf(scores[i] - mv);
    scores[i] = ex;
    atomicAdd(&denom[c * 16 + u], ex);
}

// ---- softmax pass 3: scatter x * alpha into delta_X
__global__ __launch_bounds__(256)
void scat_kernel(const int* __restrict__ ecen, const float* __restrict__ scores,
                 const float* __restrict__ denom, const float* __restrict__ xbuf,
                 float* __restrict__ dX) {
    int i = blockIdx.x * 256 + threadIdx.x;
    if (i >= EE * 16) return;
    int e = i >> 4, u = i & 15;
    int c = ecen[e];
    float alpha = scores[i] * __builtin_amdgcn_rcpf(denom[c * 16 + u] + 1e-20f);
    const float* xp = xbuf + (size_t)e * 144 + u * 9;
    float* dp = dX + (size_t)c * 144 + u * 9;
#pragma unroll
    for (int s = 0; s < 9; ++s) atomicAdd(&dp[s], xp[s] * alpha);
}

// ---- finalize h: layernorm(h + delta_h); one wave per node
__global__ __launch_bounds__(256)
void fin_h(const float* __restrict__ h, const float* __restrict__ dh,
           const float* __restrict__ gma, const float* __restrict__ bta,
           float* __restrict__ out) {
    int wave = threadIdx.x >> 5, lane = threadIdx.x & 31;
    int node = blockIdx.x * 8 + wave;
    const float* hp = h + (size_t)node * 128;
    const float* dp = dh + (size_t)node * 128;
    float x[4], sum = 0.0f, sq = 0.0f;
#pragma unroll
    for (int j = 0; j < 4; ++j) {
        x[j] = hp[lane + 32 * j] + dp[lane + 32 * j];
        sum += x[j]; sq += x[j] * x[j];
    }
#pragma unroll
    for (int off = 16; off; off >>= 1) {
        sum += __shfl_xor(sum, off, 32);
        sq  += __shfl_xor(sq,  off, 32);
    }
    float mu = sum * (1.0f / 128.0f);
    float var = sq * (1.0f / 128.0f) - mu * mu;
    float inv = rsqrtf(var + EPSV);
#pragma unroll
    for (int j = 0; j < 4; ++j) {
        int c = lane + 32 * j;
        out[(size_t)node * 128 + c] = (x[j] - mu) * inv * gma[c] + bta[c];
    }
}

// ---- finalize X: so3_layernorm(X + delta_X); one wave per node
__global__ __launch_bounds__(256)
void fin_X(const float* __restrict__ X, const float* __restrict__ dX,
           float* __restrict__ out) {
    int wave = threadIdx.x >> 5, lane = threadIdx.x & 31;
    int node = blockIdx.x * 8 + wave;
    float v[5];
    float s0 = 0.0f, s1 = 0.0f, s2 = 0.0f;
#pragma unroll
    for (int j = 0; j < 5; ++j) {
        int idx = lane + 32 * j;
        if (idx < 144) {
            float t = X[(size_t)node * 144 + idx] + dX[(size_t)node * 144 + idx];
            v[j] = t;
            int s = idx % 9;
            if (s == 0) s0 += t * t;
            else if (s < 4) s1 += t * t;
            else s2 += t * t;
        }
    }
#pragma unroll
    for (int off = 16; off; off >>= 1) {
        s0 += __shfl_xor(s0, off, 32);
        s1 += __shfl_xor(s1, off, 32);
        s2 += __shfl_xor(s2, off, 32);
    }
    float r0 = rsqrtf(s0 * (1.0f / 16.0f) + EPSV);
    float r1 = rsqrtf(s1 * (1.0f / 48.0f) + EPSV);
    float r2 = rsqrtf(s2 * (1.0f / 80.0f) + EPSV);
#pragma unroll
    for (int j = 0; j < 5; ++j) {
        int idx = lane + 32 * j;
        if (idx < 144) {
            int s = idx % 9;
            float r = (s == 0) ? r0 : (s < 4 ? r1 : r2);
            out[(size_t)node * 144 + idx] = v[j] * r;
        }
    }
}

extern "C" void kernel_launch(void* const* d_in, const int* in_sizes, int n_in,
                              void* d_out, int out_size, void* d_ws, size_t ws_size,
                              hipStream_t stream) {
    const float* h    = (const float*)d_in[0];
    const float* X    = (const float*)d_in[1];
    const float* t    = (const float*)d_in[2];
    const float* sph  = (const float*)d_in[3];
    const int*   ecen = (const int*)d_in[5];
    const int*   enbr = (const int*)d_in[6];
    const float* Wrs  = (const float*)d_in[7];
    const float* W1   = (const float*)d_in[8];
    const float* W2   = (const float*)d_in[9];
    const float* gma  = (const float*)d_in[10];
    const float* bta  = (const float*)d_in[11];
    const float* Wq   = (const float*)d_in[12];
    const float* Wk   = (const float*)d_in[13];

    char* base = (char*)d_ws;
    size_t o = 0;
    auto take = [&](size_t bytes) -> void* {
        void* p = base + o;
        o += (bytes + 255) & ~(size_t)255;
        return p;
    };
    float* g      = (float*)take((size_t)NN * GUSED * 4);
    float* Knode  = (float*)take((size_t)NN * 256 * 4);
    float* dh     = (float*)take((size_t)NN * 128 * 4);
    float* dX     = (float*)take((size_t)NN * 144 * 4);
    float* maxb   = (float*)take((size_t)NN * 16 * 4);
    float* den    = (float*)take((size_t)NN * 16 * 4);
    float* sc     = (float*)take((size_t)EE * 16 * 4);
    float* xb     = (float*)take((size_t)EE * 144 * 4);
    __bf16* Wrsc  = (__bf16*)take((size_t)GUSED * 128 * 2);
    __bf16* W1c   = (__bf16*)take((size_t)128 * 128 * 2);
    __bf16* W2c   = (__bf16*)take((size_t)GUSED * 128 * 2);
    __bf16* Wqc   = (__bf16*)take((size_t)256 * 128 * 2);
    __bf16* Wkc   = (__bf16*)take((size_t)256 * 128 * 2);

    float* out_h = (float*)d_out;
    float* out_X = out_h + (size_t)NN * 128;
    float* out_t = out_X + (size_t)NN * 144;

    // weight conversion (fp32 row-major -> bf16 col-major)
    conv_w<<<(GUSED * 128 + 255) / 256, 256, 0, stream>>>(Wrs, Wrsc, GUSED, 944);
    conv_w<<<(128 * 128 + 255) / 256, 256, 0, stream>>>(W1, W1c, 128, 128);
    conv_w<<<(GUSED * 128 + 255) / 256, 256, 0, stream>>>(W2, W2c, GUSED, 944);
    conv_w<<<(256 * 128 + 255) / 256, 256, 0, stream>>>(Wq, Wqc, 256, 256);
    conv_w<<<(256 * 128 + 255) / 256, 256, 0, stream>>>(Wk, Wkc, 256, 256);

    // init accumulators
    hipMemsetAsync(dh, 0, (size_t)NN * 128 * 4, stream);
    hipMemsetAsync(dX, 0, (size_t)NN * 144 * 4, stream);
    hipMemsetAsync(den, 0, (size_t)NN * 16 * 4, stream);
    init_max<<<(NN * 16 + 255) / 256, 256, 0, stream>>>(maxb, NN * 16);

    // per-node precompute (WMMA)
    node_kernel<<<NN / 16, 256, 0, stream>>>(h, W1c, W2c, Wkc, g, Knode);

    // per-edge heavy pass (WMMA)
    edge_kernel<<<EE / 16, 256, 0, stream>>>(t, sph, X, ecen, enbr, Wrsc, Wqc,
                                             g, Knode, dh, maxb, sc, xb);

    // scatter softmax passes
    ex_kernel<<<(EE * 16 + 255) / 256, 256, 0, stream>>>(ecen, maxb, sc, den);
    scat_kernel<<<(EE * 16 + 255) / 256, 256, 0, stream>>>(ecen, sc, den, xb, dX);

    // finalize outputs
    fin_h<<<NN / 8, 256, 0, stream>>>(h, dh, gma, bta, out_h);
    fin_X<<<NN / 8, 256, 0, stream>>>(X, dX, out_X);

    // t_ij passthrough
    hipMemcpyAsync(out_t, t, (size_t)EE * 128 * 4, hipMemcpyDeviceToDevice, stream);
}
